// MambaStack_80625126080532
// MI455X (gfx1250) — compile-verified, matrix operand order checked by reading, not run on hardware
//
#include <hip/hip_runtime.h>
#include <math.h>

// ---------------- problem constants (match reference) ----------------
#define D_MODEL   1024
#define D_STATE   64
#define D_CONV    4
#define D_INNER   2048            // EXPAND * D_MODEL
#define DT_RANK   64              // D_MODEL / 16
#define XDBL_DIM  (DT_RANK + 2*D_STATE)   // 192
#define BATCH     2
#define SEQ       4096
#define NTOK      (BATCH*SEQ)     // 8192
#define N_LAYERS  4

// ---------------- WMMA / vector types ----------------
typedef __attribute__((ext_vector_type(16))) __bf16 v16bf;
typedef __attribute__((ext_vector_type(8)))  float  v8f;
typedef __attribute__((ext_vector_type(4)))  int    i32x4;

// ---------------- GEMM tiling ----------------
#define BM    128
#define BK    32
#define LDST  40      // LDS row stride in bf16 elements: 80B = 16B-aligned, 20-dword pitch (conflict-free)

// ---------------- gfx1250 async global->LDS path (guarded) ----------------
#if __has_builtin(__builtin_amdgcn_global_load_async_to_lds_b128)
#define USE_ASYNC_LDS 1
#endif

__device__ __forceinline__ void async_b128(const void* g, void* l) {
#if defined(USE_ASYNC_LDS)
  __builtin_amdgcn_global_load_async_to_lds_b128(
      (__attribute__((address_space(1))) i32x4*)g,
      (__attribute__((address_space(3))) i32x4*)l,
      0, 0);
#else
  *(uint4*)l = *(const uint4*)g;
#endif
}

template <int N> __device__ __forceinline__ void wait_async() {
#if defined(USE_ASYNC_LDS)
#if __has_builtin(__builtin_amdgcn_s_wait_asynccnt)
  __builtin_amdgcn_s_wait_asynccnt(N);
#else
  asm volatile("s_wait_asynccnt %0" :: "i"(N) : "memory");
#endif
#endif
}

// Load one 16x32 bf16 operand fragment from an LDS tile (row-major, stride LDST),
// CDNA5 wave32 layout:
//   lanes 0-15 : row M = lane,     K in {0..7, 16..23}
//   lanes 16-31: row M = lane-16,  K in {8..15, 24..31}
__device__ __forceinline__ v16bf load_frag(const __bf16* base) {
  const int lane = threadIdx.x & 31;
  const int row  = lane & 15;
  const int khi  = lane >> 4;
  const __bf16* p = base + row * LDST + khi * 8;
  union { v16bf v; unsigned u[8]; } U;
  const unsigned* q0 = (const unsigned*)p;
  const unsigned* q1 = (const unsigned*)(p + 16);
#pragma unroll
  for (int i = 0; i < 4; ++i) U.u[i] = q0[i];
#pragma unroll
  for (int i = 0; i < 4; ++i) U.u[4 + i] = q1[i];
  return U.v;
}

// C[M,N] = A[M,K](bf16, row stride lda) * W[N,K]^T (bf16 row-major weights)
// BN_ = 64 or 128. 8 waves: 4 along M (32 rows each), 2 along N (BN_/2 each).
// OUTMODE: 0 = f32, 1 = f32 softplus(x+bias[n]), 2 = bf16 only, 3 = f32 + bf16
template <int BN_, int OUTMODE>
__global__ __launch_bounds__(256)
void wmma_gemm(float* __restrict__ Cf, __bf16* __restrict__ Cb,
               const __bf16* __restrict__ A, const __bf16* __restrict__ W,
               int K, int lda, int ldc, const float* __restrict__ bias) {
  constexpr int NF     = BN_ / 32;           // B fragments per wave
  constexpr int ACH    = (BM * 4) / 256;     // A 16B-chunks per thread (=2)
  constexpr int BCH    = (BN_ * 4) / 256;    // B 16B-chunks per thread (1 or 2)
  constexpr int NCHUNK = ACH + BCH;          // async loads in flight per tile

  __shared__ __align__(16) __bf16 As[2 * BM  * LDST];
  __shared__ __align__(16) __bf16 Bs[2 * BN_ * LDST];

  const int tid  = threadIdx.x;
  const int lane = tid & 31;
  const int wave = tid >> 5;
  const int wm   = wave & 3;       // 4 waves along M
  const int wn   = wave >> 2;      // 2 waves along N
  const int m0   = blockIdx.y * BM;
  const int n0   = blockIdx.x * BN_;
  const int nk   = K / BK;

  v8f acc[2][NF] = {};

  // issue one K-tile of async copies into buffer `buf`
  auto issue = [&](int kt, int buf) {
    __bf16* Ab = As + buf * (BM * LDST);
    __bf16* Bb = Bs + buf * (BN_ * LDST);
#pragma unroll
    for (int j = 0; j < ACH; ++j) {
      const int cid = tid + j * 256;          // 0 .. BM*4-1
      const int row = cid >> 2, c = cid & 3;  // chunk = 8 bf16 = 16B
      async_b128(A + (size_t)(m0 + row) * lda + kt * BK + c * 8,
                 Ab + row * LDST + c * 8);
    }
#pragma unroll
    for (int j = 0; j < BCH; ++j) {
      const int cid = tid + j * 256;
      const int row = cid >> 2, c = cid & 3;
      async_b128(W + (size_t)(n0 + row) * K + kt * BK + c * 8,
                 Bb + row * LDST + c * 8);
    }
  };

  issue(0, 0);
  for (int kt = 0; kt < nk; ++kt) {
    const int cur = kt & 1;
    if (kt + 1 < nk) { issue(kt + 1, cur ^ 1); wait_async<NCHUNK>(); }
    else             { wait_async<0>(); }
    __syncthreads();   // current tile resident in LDS for all waves

    const __bf16* Ab = As + cur * (BM * LDST);
    const __bf16* Bb = Bs + cur * (BN_ * LDST);
    v16bf af[2], bfr[NF];
    af[0] = load_frag(Ab + (wm * 32 + 0)  * LDST);
    af[1] = load_frag(Ab + (wm * 32 + 16) * LDST);
#pragma unroll
    for (int j = 0; j < NF; ++j)
      bfr[j] = load_frag(Bb + (wn * (BN_ / 2) + j * 16) * LDST);
#pragma unroll
    for (int i = 0; i < 2; ++i)
#pragma unroll
      for (int j = 0; j < NF; ++j)
        acc[i][j] = __builtin_amdgcn_wmma_f32_16x16x32_bf16(
            false, af[i], false, bfr[j], (short)0, acc[i][j], false, false);
    __syncthreads();   // reads done before next issue overwrites this buffer
  }

  // D layout: VGPR r -> M=r (lanes 0-15) / M=r+8 (lanes 16-31), N = lane%16
  const int cn   = lane & 15;
  const int half = lane >> 4;
#pragma unroll
  for (int i = 0; i < 2; ++i) {
#pragma unroll
    for (int j = 0; j < NF; ++j) {
      const int gn  = n0 + wn * (BN_ / 2) + j * 16 + cn;
      const int gmb = m0 + wm * 32 + i * 16 + half * 8;
      float bv = 0.0f;
      if constexpr (OUTMODE == 1) bv = bias[gn];
#pragma unroll
      for (int r = 0; r < 8; ++r) {
        float v = acc[i][j][r];
        if constexpr (OUTMODE == 1) {
          v += bv;
          v = (v > 20.0f) ? v : log1pf(expf(v));   // softplus
        }
        if constexpr (OUTMODE == 0 || OUTMODE == 1 || OUTMODE == 3)
          Cf[(size_t)(gmb + r) * ldc + gn] = v;
        if constexpr (OUTMODE == 2 || OUTMODE == 3)
          Cb[(size_t)(gmb + r) * ldc + gn] = (__bf16)v;
      }
    }
  }
}

// ---------------- f32 -> bf16 bulk convert (weights / layer-0 input) -------
__global__ __launch_bounds__(256)
void cvt_bf16_kernel(__bf16* __restrict__ dst, const float* __restrict__ src) {
  const long i = (long)blockIdx.x * 256 + threadIdx.x;
  dst[i] = (__bf16)src[i];
}

// ---------------- depthwise causal conv (width 4) + SiLU -------------------
__global__ __launch_bounds__(256)
void conv_silu_kernel(float* __restrict__ xa, __bf16* __restrict__ xab,
                      const float* __restrict__ xz,
                      const float* __restrict__ cw, const float* __restrict__ cb) {
  const long idx = (long)blockIdx.x * 256 + threadIdx.x;   // over NTOK*D_INNER
  const int  d   = (int)(idx % D_INNER);
  const long tok = idx / D_INNER;
  const int  l   = (int)(tok % SEQ);
  float acc = cb[d];
#pragma unroll
  for (int k = 0; k < D_CONV; ++k) {
    const int ll = l - (D_CONV - 1) + k;
    if (ll >= 0)
      acc += xz[(tok + (ll - l)) * (2 * D_INNER) + d] * cw[d * D_CONV + k];
  }
  const float s = acc / (1.0f + expf(-acc));
  xa[idx]  = s;
  xab[idx] = (__bf16)s;
}

// ---------------- selective scan: one wave32 per (b, channel) --------------
__global__ __launch_bounds__(256)
void scan_kernel(float* __restrict__ y, const float* __restrict__ dt,
                 const float* __restrict__ xa, const float* __restrict__ xdbl,
                 const float* __restrict__ A_log) {
  const int w    = blockIdx.x * 8 + (threadIdx.x >> 5);
  const int lane = threadIdx.x & 31;
  const int b    = w / D_INNER;
  const int d    = w % D_INNER;
  const int s0   = lane, s1 = lane + 32;
  const float a0 = -expf(A_log[d * D_STATE + s0]);
  const float a1 = -expf(A_log[d * D_STATE + s1]);
  float h0 = 0.0f, h1 = 0.0f;
  const long tokbase = (long)b * SEQ;
  for (int t = 0; t < SEQ; ++t) {
    const long tok = tokbase + t;
    const float dtv = dt[tok * D_INNER + d];
    const float xv  = xa[tok * D_INNER + d];
    const float* xd = xdbl + tok * XDBL_DIM;
    const float B0 = xd[DT_RANK + s0];
    const float B1 = xd[DT_RANK + s1];
    const float C0 = xd[DT_RANK + D_STATE + s0];
    const float C1 = xd[DT_RANK + D_STATE + s1];
    const float u = dtv * xv;
    h0 = h0 * expf(dtv * a0) + u * B0;
    h1 = h1 * expf(dtv * a1) + u * B1;
    float p = h0 * C0 + h1 * C1;
#pragma unroll
    for (int off = 16; off >= 1; off >>= 1) p += __shfl_xor(p, off, 32);
    if (lane == 0) y[tok * D_INNER + d] = p;
  }
}

// ---------------- y = (y + xa*D) * silu(z); writes f32 (in place) + bf16 ---
__global__ __launch_bounds__(256)
void gate_kernel(float* __restrict__ y, __bf16* __restrict__ yb,
                 const float* __restrict__ xa, const float* __restrict__ xz,
                 const float* __restrict__ Dp) {
  const long idx = (long)blockIdx.x * 256 + threadIdx.x;
  const int  d   = (int)(idx % D_INNER);
  const long tok = idx / D_INNER;
  const float z  = xz[tok * (2 * D_INNER) + D_INNER + d];
  const float sz = z / (1.0f + expf(-z));
  const float v  = (y[idx] + xa[idx] * Dp[d]) * sz;
  y[idx]  = v;
  yb[idx] = (__bf16)v;
}

// ---------------- host launcher ----------------
extern "C" void kernel_launch(void* const* d_in, const int* in_sizes, int n_in,
                              void* d_out, int out_size, void* d_ws, size_t ws_size,
                              hipStream_t stream) {
  (void)in_sizes; (void)n_in; (void)out_size; (void)ws_size;
  const float* x      = (const float*)d_in[0];
  const float* in_w   = (const float*)d_in[1];   // (4, 4096, 1024)
  const float* conv_w = (const float*)d_in[2];   // (4, 2048, 4)
  const float* conv_b = (const float*)d_in[3];   // (4, 2048)
  const float* xp_w   = (const float*)d_in[4];   // (4, 192, 2048)
  const float* dt_w   = (const float*)d_in[5];   // (4, 2048, 64)
  const float* dt_b   = (const float*)d_in[6];   // (4, 2048)
  const float* A_log  = (const float*)d_in[7];   // (4, 2048, 64)
  const float* Dp     = (const float*)d_in[8];   // (4, 2048)
  const float* out_w  = (const float*)d_in[9];   // (4, 1024, 2048)
  float* out = (float*)d_out;

  // ---- workspace carve: f32 buffers first, then bf16 ----
  float* wsf  = (float*)d_ws;
  float* xz   = wsf;  wsf += (size_t)NTOK * 2 * D_INNER;   // 8192 x 4096
  float* xa   = wsf;  wsf += (size_t)NTOK * D_INNER;
  float* xdbl = wsf;  wsf += (size_t)NTOK * XDBL_DIM;
  float* dt   = wsf;  wsf += (size_t)NTOK * D_INNER;
  float* yb   = wsf;  wsf += (size_t)NTOK * D_INNER;
  __bf16* wsb    = (__bf16*)wsf;
  __bf16* xbf    = wsb;  wsb += (size_t)NTOK * D_MODEL;    // layer input (bf16)
  __bf16* xa_bf  = wsb;  wsb += (size_t)NTOK * D_INNER;
  __bf16* xd_bf  = wsb;  wsb += (size_t)NTOK * XDBL_DIM;
  __bf16* yb_bf  = wsb;  wsb += (size_t)NTOK * D_INNER;
  __bf16* w_in   = wsb;  wsb += (size_t)N_LAYERS * 2 * D_INNER * D_MODEL;
  __bf16* w_xp   = wsb;  wsb += (size_t)N_LAYERS * XDBL_DIM * D_INNER;
  __bf16* w_dt   = wsb;  wsb += (size_t)N_LAYERS * D_INNER * DT_RANK;
  __bf16* w_out  = wsb;  wsb += (size_t)N_LAYERS * D_MODEL * D_INNER;

  const dim3 blk(256);
  const long nelem = (long)NTOK * D_INNER;                 // 16,777,216

  // ---- one-time (per launch) bf16 conversions ----
  cvt_bf16_kernel<<<(unsigned)(((size_t)NTOK * D_MODEL) / 256), blk, 0, stream>>>(xbf, x);
  cvt_bf16_kernel<<<(unsigned)(((size_t)N_LAYERS * 2 * D_INNER * D_MODEL) / 256), blk, 0, stream>>>(w_in, in_w);
  cvt_bf16_kernel<<<(unsigned)(((size_t)N_LAYERS * XDBL_DIM * D_INNER) / 256), blk, 0, stream>>>(w_xp, xp_w);
  cvt_bf16_kernel<<<(unsigned)(((size_t)N_LAYERS * D_INNER * DT_RANK) / 256), blk, 0, stream>>>(w_dt, dt_w);
  cvt_bf16_kernel<<<(unsigned)(((size_t)N_LAYERS * D_MODEL * D_INNER) / 256), blk, 0, stream>>>(w_out, out_w);

  for (int i = 0; i < N_LAYERS; ++i) {
    // 1) in_proj: xz(f32) = xbf @ in_w^T   (8192 x 4096, K=1024)
    wmma_gemm<128, 0><<<dim3(2 * D_INNER / 128, NTOK / BM), blk, 0, stream>>>(
        xz, nullptr, xbf, w_in + (size_t)i * 2 * D_INNER * D_MODEL,
        D_MODEL, D_MODEL, 2 * D_INNER, nullptr);

    // 2) causal depthwise conv + SiLU -> xa (f32 + bf16)
    conv_silu_kernel<<<(unsigned)(nelem / 256), blk, 0, stream>>>(
        xa, xa_bf, xz, conv_w + (size_t)i * D_INNER * D_CONV,
        conv_b + (size_t)i * D_INNER);

    // 3) x_proj: xdbl(f32 + bf16) = xa_bf @ xp_w^T  (8192 x 192, K=2048)
    wmma_gemm<64, 3><<<dim3(XDBL_DIM / 64, NTOK / BM), blk, 0, stream>>>(
        xdbl, xd_bf, xa_bf, w_xp + (size_t)i * XDBL_DIM * D_INNER,
        D_INNER, D_INNER, XDBL_DIM, nullptr);

    // 4) dt(f32) = softplus(xd_bf[:, :64] @ dt_w^T + dt_b)  (8192 x 2048, K=64)
    wmma_gemm<128, 1><<<dim3(D_INNER / 128, NTOK / BM), blk, 0, stream>>>(
        dt, nullptr, xd_bf, w_dt + (size_t)i * D_INNER * DT_RANK,
        DT_RANK, XDBL_DIM, D_INNER, dt_b + (size_t)i * D_INNER);

    // 5) selective scan -> yb (f32)
    scan_kernel<<<(BATCH * D_INNER) / 8, blk, 0, stream>>>(
        yb, dt, xa, xdbl, A_log + (size_t)i * D_INNER * D_STATE);

    // 6) gate: yb = (yb + xa*D) * silu(z), also write bf16
    gate_kernel<<<(unsigned)(nelem / 256), blk, 0, stream>>>(
        yb, yb_bf, xa, xz, Dp + (size_t)i * D_INNER);

    // 7) out_proj: layers 0-2 write bf16 residual (next in_proj input);
    //    last layer writes f32 d_out.
    if (i < N_LAYERS - 1) {
      wmma_gemm<128, 2><<<dim3(D_MODEL / 128, NTOK / BM), blk, 0, stream>>>(
          nullptr, xbf, yb_bf, w_out + (size_t)i * D_MODEL * D_INNER,
          D_INNER, D_INNER, D_MODEL, nullptr);
    } else {
      wmma_gemm<128, 0><<<dim3(D_MODEL / 128, NTOK / BM), blk, 0, stream>>>(
          out, nullptr, yb_bf, w_out + (size_t)i * D_MODEL * D_INNER,
          D_INNER, D_INNER, D_MODEL, nullptr);
    }
  }
}
